// DCNv3_44409961840897
// MI455X (gfx1250) — matrix operand order repaired and checked by reading, not compiled, not run on hardware
//
#include <hip/hip_runtime.h>

typedef __attribute__((ext_vector_type(16))) __bf16       v16bf;
typedef __attribute__((ext_vector_type(8)))  float        v8f;
typedef __attribute__((ext_vector_type(8)))  unsigned int v8u;
typedef __attribute__((ext_vector_type(4)))  unsigned int v4u;

#define BATCH   4
#define C_IN    128
#define C_OUT   128
#define HH      64
#define WW      64
#define NPIX    4096          // 64*64
#define TAPS    9
#define KDIM    1152          // 128*9
#define KK_TILES 36           // 1152/32
#define NTILES  256           // 4096/16
#define FRAG_DW 256           // dwords per 16x32 bf16 fragment (32 lanes * 8 dw)
#define TILE_DW (KK_TILES*FRAG_DW)   // 9216 dwords per (b, ntile)

// ---- workspace layout (dword offsets) ----
#define WS_A_MAIN 0
#define WS_A_OM   (8*KK_TILES*FRAG_DW)              // 73728
#define WS_OFF    (WS_A_OM + 2*KK_TILES*FRAG_DW)    // 92160
#define WS_MASK   (WS_OFF + BATCH*18*NPIX)          // 387072
#define WS_BFRAG  (WS_MASK + BATCH*9*NPIX)          // 534528
#define WS_TOTAL_DW (WS_BFRAG + BATCH*NTILES*TILE_DW)  // ~9.97M dw ~= 40 MB

__device__ __forceinline__ unsigned short f2bf(float f) {
    unsigned int u = __float_as_uint(f);
    unsigned int r = u + 0x7FFFu + ((u >> 16) & 1u);   // round-to-nearest-even
    return (unsigned short)(r >> 16);
}

__device__ __forceinline__ int iclamp(int v, int lo, int hi) {
    return v < lo ? lo : (v > hi ? hi : v);
}

// ---------------------------------------------------------------------------
// 1) Pack main + offset/mask conv weights into WMMA A-fragment bf16 layout.
//    A 16x32 bf16 fragment: lane L: M = L%16, khalf = L/16;
//    dword v (0..7): Klocal = 16*khalf + 2v (+p).
//    Stored so each lane's 8 dwords are contiguous (32B per lane).
// ---------------------------------------------------------------------------
__global__ __launch_bounds__(256)
void pack_w_kernel(const float* __restrict__ wmain,
                   const float* __restrict__ offw,
                   const float* __restrict__ maskw,
                   unsigned int* __restrict__ ws) {
    const int MAIN_DW = 8 * KK_TILES * FRAG_DW;
    const int TOT     = MAIN_DW + 2 * KK_TILES * FRAG_DW;
    int d = blockIdx.x * 256 + threadIdx.x;
    if (d >= TOT) return;
    bool is_om = (d >= MAIN_DW);
    int dl  = is_om ? (d - MAIN_DW) : d;
    int mt  = dl / (KK_TILES * FRAG_DW);
    int rem = dl % (KK_TILES * FRAG_DW);
    int kk  = rem / FRAG_DW;
    int l8  = rem % FRAG_DW;
    int lane = l8 >> 3;
    int v    = l8 & 7;
    int khalf = lane >> 4;
    int M     = mt * 16 + (lane & 15);
    unsigned int out = 0;
    #pragma unroll
    for (int p = 0; p < 2; ++p) {
        int base = 2 * v;
        int K = kk * 32 + base + khalf * 16 + p;        // < 1152
        int c = K / TAPS, t = K % TAPS;
        float w;
        if (!is_om)        w = wmain[(M * C_IN + c) * TAPS + t];
        else if (M < 18)   w = offw [(M * C_IN + c) * TAPS + t];
        else if (M < 27)   w = maskw[((M - 18) * C_IN + c) * TAPS + t];
        else               w = 0.f;
        out |= (unsigned int)f2bf(w) << (16 * p);
    }
    ws[(is_om ? WS_A_OM : WS_A_MAIN) + dl] = out;
}

// ---------------------------------------------------------------------------
// 2) im2col of x into bf16 WMMA B-fragments (for the offset/mask conv GEMM).
//    B 32x16 bf16 fragment: lane L: N = L%16, khalf = L/16;
//    dword v: Klocal = 16*khalf + 2v (+p).
//    One block per (batch, 16-pixel tile); LDS-staged, coalesced b128 flush.
// ---------------------------------------------------------------------------
__global__ __launch_bounds__(256)
void im2col_kernel(const float* __restrict__ x, unsigned int* __restrict__ ws) {
    __shared__ __align__(16) unsigned short sfrag16[KK_TILES * 512]; // 36 KB
    int gt = blockIdx.x;                  // 0..1023
    int b = gt >> 8, nt = gt & 255;
    int tid = threadIdx.x;
    int j = tid & 15, w = tid >> 4;
    int npix = nt * 16 + j;
    int yo = npix >> 6, xo = npix & 63;
    const float* xb = x + (size_t)b * C_IN * NPIX;
    #pragma unroll
    for (int ci = 0; ci < 8; ++ci) {
        int c = w * 8 + ci;
        const float* xc = xb + c * NPIX;
        #pragma unroll
        for (int t = 0; t < TAPS; ++t) {
            int yy = yo - 1 + t / 3;
            int xx = xo - 1 + t % 3;
            float val = (yy >= 0 && yy < HH && xx >= 0 && xx < WW)
                        ? xc[yy * WW + xx] : 0.f;
            int k = c * TAPS + t;
            int kk = k >> 5, kl = k & 31;
            int lane = (kl >> 4) * 16 + j;
            int vv = (kl & 15) >> 1, p = kl & 1;
            sfrag16[kk * 512 + lane * 16 + vv * 2 + p] = f2bf(val);
        }
    }
    __syncthreads();
    v4u* dst = (v4u*)(ws + WS_BFRAG + (size_t)gt * TILE_DW);
    const v4u* src = (const v4u*)sfrag16;
    for (int i = tid; i < TILE_DW / 4; i += 256)
        __builtin_nontemporal_store(src[i], &dst[i]);   // streamed, no reuse here
}

// ---------------------------------------------------------------------------
// 3) Offset/mask GEMM: M=32 (27 used), K=1152, N=16384 via WMMA bf16.
//    One wave per 16-pixel N-tile, 2 accumulators. Epilogue: bias, sigmoid
//    on mask rows, scatter to f32 offset / mask planes in workspace.
// ---------------------------------------------------------------------------
__global__ __launch_bounds__(256)
void gemm_om_kernel(const unsigned int* __restrict__ ws,
                    const float* __restrict__ offb,
                    const float* __restrict__ maskb,
                    float* __restrict__ wsf) {
    int wid = threadIdx.x >> 5;
    int lane = threadIdx.x & 31;
    int gt = blockIdx.x * 8 + wid;        // 0..1023
    int b = gt >> 8, nt = gt & 255;
    const unsigned int* bbase = ws + WS_BFRAG + (size_t)gt * TILE_DW + lane * 8;
    const unsigned int* abase = ws + WS_A_OM + lane * 8;
    v8f acc0 = {}; v8f acc1 = {};
    for (int kk = 0; kk < KK_TILES; ++kk) {
        v8u bu = __builtin_nontemporal_load((const v8u*)(bbase + kk * FRAG_DW));
        v16bf bvec = __builtin_bit_cast(v16bf, bu);
        v16bf a0   = __builtin_bit_cast(v16bf, *(const v8u*)(abase + kk * FRAG_DW));
        v16bf a1   = __builtin_bit_cast(v16bf, *(const v8u*)(abase + (KK_TILES + kk) * FRAG_DW));
        acc0 = __builtin_amdgcn_wmma_f32_16x16x32_bf16(false, a0, false, bvec,
                                                       (short)0, acc0, false, false);
        acc1 = __builtin_amdgcn_wmma_f32_16x16x32_bf16(false, a1, false, bvec,
                                                       (short)0, acc1, false, false);
    }
    int n = nt * 16 + (lane & 15);
    int mofs = (lane >> 4) * 8;           // C/D layout: VGPR r -> M = r (+8 hi half)
    #pragma unroll
    for (int r = 0; r < 8; ++r) {
        int row0 = r + mofs;              // 0..15 : all offset rows
        wsf[WS_OFF + ((size_t)b * 18 + row0) * NPIX + n] = acc0[r] + offb[row0];
        int row1 = 16 + r + mofs;         // 16..31
        float v1 = acc1[r];
        if (row1 < 18) {
            wsf[WS_OFF + ((size_t)b * 18 + row1) * NPIX + n] = v1 + offb[row1];
        } else if (row1 < 27) {
            float z = v1 + maskb[row1 - 18];
            wsf[WS_MASK + ((size_t)b * 9 + (row1 - 18)) * NPIX + n] =
                1.f / (1.f + __expf(-z));
        }
    }
}

// ---------------------------------------------------------------------------
// 4) Modulated bilinear sampling -> bf16 B-fragments (overwrites im2col buf).
//    Phase A: 144 threads precompute per-(tap,pixel) corner indices and
//    bilinear-weights*mask. Phase B: gather x (L2-resident), pack fragments.
// ---------------------------------------------------------------------------
__global__ __launch_bounds__(256)
void sample_kernel(const float* __restrict__ x,
                   const float* __restrict__ wsf,
                   unsigned int* __restrict__ ws) {
    __shared__ __align__(16) unsigned short sfrag16[KK_TILES * 512]; // 36 KB
    __shared__ float sw[TAPS * 16][4];
    __shared__ int   sidx[TAPS * 16][4];
    int gt = blockIdx.x;
    int b = gt >> 8, nt = gt & 255;
    int tid = threadIdx.x;
    if (tid < TAPS * 16) {
        int j = tid & 15, t = tid >> 4;
        int npix = nt * 16 + j;
        int yo = npix >> 6, xo = npix & 63;
        float offy = wsf[WS_OFF + ((size_t)b * 18 + 2 * t) * NPIX + npix];
        float offx = wsf[WS_OFF + ((size_t)b * 18 + 2 * t + 1) * NPIX + npix];
        float m    = wsf[WS_MASK + ((size_t)b * 9 + t) * NPIX + npix];
        float py = (float)(yo - 1 + t / 3) + offy;
        float px = (float)(xo - 1 + t % 3) + offx;
        float fy = floorf(py), fx = floorf(px);
        float ly = py - fy, lx = px - fx;
        int y0 = (int)fy, x0 = (int)fx;
        int y1 = y0 + 1, x1 = x0 + 1;
        float vy0 = (y0 >= 0 && y0 < HH) ? 1.f : 0.f;
        float vy1 = (y1 >= 0 && y1 < HH) ? 1.f : 0.f;
        float vx0 = (x0 >= 0 && x0 < WW) ? 1.f : 0.f;
        float vx1 = (x1 >= 0 && x1 < WW) ? 1.f : 0.f;
        int y0c = iclamp(y0, 0, HH - 1), y1c = iclamp(y1, 0, HH - 1);
        int x0c = iclamp(x0, 0, WW - 1), x1c = iclamp(x1, 0, WW - 1);
        sw[tid][0] = (1.f - ly) * (1.f - lx) * vy0 * vx0 * m;
        sw[tid][1] = (1.f - ly) * lx         * vy0 * vx1 * m;
        sw[tid][2] = ly         * (1.f - lx) * vy1 * vx0 * m;
        sw[tid][3] = ly         * lx         * vy1 * vx1 * m;
        sidx[tid][0] = y0c * WW + x0c;
        sidx[tid][1] = y0c * WW + x1c;
        sidx[tid][2] = y1c * WW + x0c;
        sidx[tid][3] = y1c * WW + x1c;
    }
    __syncthreads();
    int j = tid & 15, w = tid >> 4;
    const float* xb = x + (size_t)b * C_IN * NPIX;
    #pragma unroll
    for (int ci = 0; ci < 8; ++ci) {
        int c = w * 8 + ci;
        const float* xc = xb + c * NPIX;
        #pragma unroll
        for (int t = 0; t < TAPS; ++t) {
            int s = t * 16 + j;
            float val = sw[s][0] * xc[sidx[s][0]] + sw[s][1] * xc[sidx[s][1]]
                      + sw[s][2] * xc[sidx[s][2]] + sw[s][3] * xc[sidx[s][3]];
            int k = c * TAPS + t;
            int kk = k >> 5, kl = k & 31;
            int lane = (kl >> 4) * 16 + j;
            int vv = (kl & 15) >> 1, p = kl & 1;
            sfrag16[kk * 512 + lane * 16 + vv * 2 + p] = f2bf(val);
        }
    }
    __syncthreads();
    v4u* dst = (v4u*)(ws + WS_BFRAG + (size_t)gt * TILE_DW);
    const v4u* src = (const v4u*)sfrag16;
    for (int i = tid; i < TILE_DW / 4; i += 256)
        __builtin_nontemporal_store(src[i], &dst[i]);
}

// ---------------------------------------------------------------------------
// 5) Main GEMM: M=128, K=1152, N=16384. One wave per 16-pixel N-tile,
//    8 WMMA accumulators (16Nx128M). The 8 waves of a block share all 8
//    A fragments per K-step -> stage A through a double-buffered LDS tile
//    (8 KB/buffer): cooperative global load of A(kk+1) overlaps the 8 WMMAs
//    on A(kk) read from LDS. Cuts global A traffic 8x; B streams NT.
//    Fused BN + SiLU epilogue.
// ---------------------------------------------------------------------------
__global__ __launch_bounds__(256)
void gemm_main_kernel(const unsigned int* __restrict__ ws,
                      const float* __restrict__ gamma,
                      const float* __restrict__ beta,
                      const float* __restrict__ mean,
                      const float* __restrict__ var,
                      float* __restrict__ out) {
    __shared__ __align__(32) unsigned int sA[2][8 * FRAG_DW];   // 2 x 8 KB
    int tid = threadIdx.x;
    int wid = tid >> 5, lane = tid & 31;
    int gt = blockIdx.x * 8 + wid;
    int b = gt >> 8, nt = gt & 255;
    const unsigned int* bbase = ws + WS_BFRAG + (size_t)gt * TILE_DW + lane * 8;

    // cooperative A staging: thread tid owns (mt = tid>>5, lane8 = (tid&31)*8)
    int amt   = tid >> 5;
    int al8   = (tid & 31) * 8;
    const unsigned int* agbase = ws + WS_A_MAIN + (size_t)amt * KK_TILES * FRAG_DW + al8;
    int sidxA = amt * FRAG_DW + al8;

    v8f acc[8] = {};

    // preload kk = 0 into buffer 0
    {
        v8u r0 = *(const v8u*)agbase;
        *(v8u*)&sA[0][sidxA] = r0;
    }
    __syncthreads();

    for (int kk = 0; kk < KK_TILES; ++kk) {
        int p = kk & 1;
        // start fetching next A tile from global (hidden under WMMA work)
        v8u rn = {};
        if (kk + 1 < KK_TILES)
            rn = *(const v8u*)(agbase + (size_t)(kk + 1) * FRAG_DW);
        // B fragment: consumed exactly once -> non-temporal
        v8u bu = __builtin_nontemporal_load((const v8u*)(bbase + kk * FRAG_DW));
        v16bf bvec = __builtin_bit_cast(v16bf, bu);
        #pragma unroll
        for (int mt = 0; mt < 8; ++mt) {
            v8u au = *(const v8u*)&sA[p][mt * FRAG_DW + lane * 8];
            v16bf avec = __builtin_bit_cast(v16bf, au);
            acc[mt] = __builtin_amdgcn_wmma_f32_16x16x32_bf16(
                false, avec, false, bvec, (short)0, acc[mt], false, false);
        }
        if (kk + 1 < KK_TILES)
            *(v8u*)&sA[p ^ 1][sidxA] = rn;
        __syncthreads();
    }

    int n = nt * 16 + (lane & 15);
    int mofs = (lane >> 4) * 8;
    float* outb = out + (size_t)b * C_OUT * NPIX + n;
    #pragma unroll
    for (int mt = 0; mt < 8; ++mt) {
        #pragma unroll
        for (int r = 0; r < 8; ++r) {
            int o = mt * 16 + r + mofs;
            float g = acc[mt][r];
            float sc = gamma[o] * __frsqrt_rn(var[o] + 1e-5f);
            float z = (g - mean[o]) * sc + beta[o];
            __builtin_nontemporal_store(z * (1.f / (1.f + __expf(-z))),
                                        &outb[(size_t)o * NPIX]);
        }
    }
}

// ---------------------------------------------------------------------------
extern "C" void kernel_launch(void* const* d_in, const int* in_sizes, int n_in,
                              void* d_out, int out_size, void* d_ws, size_t ws_size,
                              hipStream_t stream) {
    const float* x     = (const float*)d_in[0];
    const float* offw  = (const float*)d_in[1];
    const float* offb  = (const float*)d_in[2];
    const float* maskw = (const float*)d_in[3];
    const float* maskb = (const float*)d_in[4];
    const float* wmain = (const float*)d_in[5];
    const float* gamma = (const float*)d_in[6];
    const float* beta  = (const float*)d_in[7];
    const float* mean  = (const float*)d_in[8];
    const float* var   = (const float*)d_in[9];
    unsigned int* ws   = (unsigned int*)d_ws;
    float*        wsf  = (float*)d_ws;
    float*        out  = (float*)d_out;

    if (ws_size < (size_t)WS_TOTAL_DW * 4) return;   // ~40 MB workspace needed

    pack_w_kernel   <<<360,  256, 0, stream>>>(wmain, offw, maskw, ws);
    im2col_kernel   <<<1024, 256, 0, stream>>>(x, ws);
    gemm_om_kernel  <<<128,  256, 0, stream>>>(ws, offb, maskb, wsf);
    sample_kernel   <<<1024, 256, 0, stream>>>(x, wsf, ws);
    gemm_main_kernel<<<128,  256, 0, stream>>>(ws, gamma, beta, mean, var, out);
}
